// ResNetEncoder_67843303407688
// MI455X (gfx1250) — compile-verified, matrix operand order checked by reading, not run on hardware
//
#include <hip/hip_runtime.h>
#include <hip/hip_bf16.h>

// ---------------------------------------------------------------------------
// CDNA5 (gfx1250) spherical ResNet encoder.
//   - gather+GEMM fused, bf16 WMMA (v_wmma_f32_16x16x32_bf16), f32 accumulate
//   - 32x64 block tile, 4 wave32, 2 WMMA per wave per K-step
//   - LDS fragments read as ds_load_b128 (A row-major, W transposed), double-
//     buffered, one barrier per K-step
//   - packed f32->bf16 conversion via vector convert (v_cvt_pk_bf16_f32)
//   - BN statistics fused into the GEMM epilogue (cross-lane shfl + atomics)
//   - conv bias folded away (cancels under training-mode BatchNorm)
// ---------------------------------------------------------------------------

typedef __attribute__((ext_vector_type(16))) __bf16 v16bf;
typedef __attribute__((ext_vector_type(8)))  __bf16 v8bf;
typedef __attribute__((ext_vector_type(2)))  __bf16 v2bf;
typedef __attribute__((ext_vector_type(8)))  float  v8f;
typedef __attribute__((ext_vector_type(2)))  float  v2f;

#define NEG_SLOPE 0.2f
#define BN_EPS    1e-5f

// packed pair conversion: lowers to v_cvt_pk_bf16_f32 on gfx1250
static __device__ __forceinline__ unsigned pack2(float a, float b) {
    v2f  f = {a, b};
    v2bf h = __builtin_convertvector(f, v2bf);
    return __builtin_bit_cast(unsigned, h);
}
static __device__ __forceinline__ v16bf combine(v8bf lo, v8bf hi) {
    return __builtin_shufflevector(lo, hi, 0,1,2,3,4,5,6,7,8,9,10,11,12,13,14,15);
}

// ------------------------------ small helpers ------------------------------

__global__ void zero_kernel(float* __restrict__ p, int n) {
    int i = blockIdx.x * blockDim.x + threadIdx.x;
    if (i < n) p[i] = 0.f;
}

// per-channel sum/sumsq for the conv1 path only (C == 32)
__global__ __launch_bounds__(256)
void stats32_kernel(const float* __restrict__ Y, long M, float* __restrict__ stats) {
    int  t = threadIdx.x;
    int  c = t & 31;
    long r      = (long)blockIdx.x * 8 + (t >> 5);
    long stride = (long)gridDim.x * 8;
    float s = 0.f, q = 0.f;
    for (; r < M; r += stride) {
        float v = Y[(r << 5) + c];
        s += v; q += v * v;
    }
    atomicAdd(&stats[c],      s);
    atomicAdd(&stats[32 + c], q);
}

// in-place BN (+optional residual with channel doubling) + LeakyReLU, float4
__global__ __launch_bounds__(256)
void apply_bn_kernel(float* __restrict__ Y, long M, int cqShift,
                     const float* __restrict__ gamma,
                     const float* __restrict__ beta,
                     const float* __restrict__ stats,
                     const float* __restrict__ res, int resC) {
    const int Cq = 1 << cqShift;          // C / 4
    const int C  = Cq << 2;
    long idx   = (long)blockIdx.x * blockDim.x + threadIdx.x;
    long total = M << cqShift;
    if (idx >= total) return;
    int  cq = (int)(idx & (Cq - 1));
    long r  = idx >> cqShift;
    int  cb = cq << 2;
    float rvv[4] = {0.f, 0.f, 0.f, 0.f};
    if (res) {
        const float* rp = res + r * (long)resC + (cb & (resC - 1)); // concat wrap
        rvv[0] = rp[0]; rvv[1] = rp[1]; rvv[2] = rp[2]; rvv[3] = rp[3];
    }
    float4 y = ((const float4*)Y)[idx];
    float* yv = (float*)&y;
    const float cnt = (float)M;
#pragma unroll
    for (int k = 0; k < 4; ++k) {
        int   c    = cb + k;
        float mean = stats[c] / cnt;
        float var  = stats[C + c] / cnt - mean * mean;
        float v = (yv[k] - mean) * rsqrtf(var + BN_EPS) * gamma[c] + beta[c];
        v += rvv[k];
        yv[k] = v > 0.f ? v : NEG_SLOPE * v;
    }
    ((float4*)Y)[idx] = y;
}

// mean over one-ring of the first n2 coarse vertices (float4 per thread)
__global__ __launch_bounds__(256)
void pool_kernel(const float* __restrict__ X, const int* __restrict__ no,
                 float* __restrict__ O, int Nf, int n2, int cqShift) {
    const int Cq = 1 << cqShift;          // C / 4
    long idx   = (long)blockIdx.x * blockDim.x + threadIdx.x;
    long total = (64L * n2) << cqShift;
    if (idx >= total) return;
    int  cq = (int)(idx & (Cq - 1));
    long t2 = idx >> cqShift;
    int  n  = (int)(t2 % n2);
    int  b  = (int)(t2 / n2);
    float4 s = {0.f, 0.f, 0.f, 0.f};
#pragma unroll
    for (int j = 0; j < 7; ++j) {
        int nbr = no[n * 7 + j];
        const float4 xv = *(const float4*)(X + (((long)b * Nf + nbr) << (cqShift + 2)) + (cq << 2));
        s.x += xv.x; s.y += xv.y; s.z += xv.z; s.w += xv.w;
    }
    const float inv7 = 1.f / 7.f;
    s.x *= inv7; s.y *= inv7; s.z *= inv7; s.w *= inv7;
    ((float4*)O)[idx] = s;
}

// first conv: K = 7*3 = 21, Cout = 32 -> VALU FMA kernel (bias folded)
__global__ __launch_bounds__(256)
void conv1_kernel(const float* __restrict__ X, const int* __restrict__ no,
                  const float* __restrict__ W, float* __restrict__ Y, int Nv) {
    __shared__ float Ws[21 * 32];
    for (int i = threadIdx.x; i < 21 * 32; i += blockDim.x) Ws[i] = W[i];
    __syncthreads();
    long row = (long)blockIdx.x * blockDim.x + threadIdx.x;
    long M   = 64L * Nv;
    if (row >= M) return;
    int b = (int)(row / Nv), v = (int)(row % Nv);
    float xin[21];
#pragma unroll
    for (int j = 0; j < 7; ++j) {
        int nbr = no[v * 7 + j];
        const float* p = X + ((long)b * Nv + nbr) * 3;
        xin[j * 3 + 0] = p[0]; xin[j * 3 + 1] = p[1]; xin[j * 3 + 2] = p[2];
    }
    float acc[32];
#pragma unroll
    for (int o = 0; o < 32; ++o) acc[o] = 0.f;
#pragma unroll
    for (int k = 0; k < 21; ++k) {
        float xv = xin[k];
#pragma unroll
        for (int o = 0; o < 32; ++o) acc[o] = fmaf(xv, Ws[k * 32 + o], acc[o]);
    }
    float4* yp = (float4*)(Y + (row << 5));
#pragma unroll
    for (int o = 0; o < 8; ++o)
        yp[o] = make_float4(acc[4*o], acc[4*o+1], acc[4*o+2], acc[4*o+3]);
}

// ------------------------- fused gather + WMMA GEMM ------------------------
// Y[b*Nv+v, n] = sum_k A[row,k] * W[k,n],  A[row, j*Cin+c] = X[b, no[v*7+j], c]
// Cin multiple of 32 -> each 32-wide K step is ONE neighbor's contiguous slab.
// Block: 128 thr (4 wave32), tile 32 rows x 64 cols; 2 accumulators per wave
// share one B fragment. BN sum/sumsq fused into the epilogue.
template <int Cin, int Cout>
__global__ __launch_bounds__(128)
void conv_wmma_kernel(const float* __restrict__ X, const int* __restrict__ no,
                      const float* __restrict__ W, float* __restrict__ Y,
                      int Nv, float* __restrict__ stats) {
    constexpr int KS = (7 * Cin) / 32;
    // stride 40 ushorts = 80B: 16B-aligned rows, conflict-free b128 reads
    __shared__ __align__(16) unsigned short As[2][32][40]; // A tile (row-major K)
    __shared__ __align__(16) unsigned short Wt[2][64][40]; // W tile transposed [n][k]
    __shared__ int s_nbr[32][7];
    __shared__ int s_base[32];      // b * Nv per row

    const int t      = threadIdx.x;
    const int lane   = t & 31;
    const int wave   = t >> 5;
    const int row0   = blockIdx.x * 32;
    const int colBlk = blockIdx.y * 64;

    if (t < 32) s_base[t] = ((row0 + t) / Nv) * Nv;
    for (int i = t; i < 224; i += 128) {
        int r = i / 7, j = i % 7;
        s_nbr[r][j] = no[((row0 + r) % Nv) * 7 + j];
    }
    __syncthreads();

    auto stage = [&](int s, int buf) {
        const int j  = (s * 32) / Cin;      // neighbor for this K slab
        const int c0 = (s * 32) % Cin;      // contiguous 32-channel base
        // A: 32 rows x 16 dword-pairs, 4 per thread (coalesced float2 reads)
#pragma unroll
        for (int i = 0; i < 4; ++i) {
            int e = t + i * 128;
            int r = e >> 4, cp = e & 15;
            const float* px = X + (long)(s_base[r] + s_nbr[r][j]) * Cin + c0 + 2 * cp;
            ((unsigned*)As[buf][r])[cp] = pack2(px[0], px[1]);
        }
        // W transposed: 64 cols x 16 k-pairs, 8 per thread (coalesced over n)
#pragma unroll
        for (int i = 0; i < 8; ++i) {
            int e = t + i * 128;
            int n = e & 63, kp = e >> 6;
            const float* pw = W + (long)(s * 32 + 2 * kp) * Cout + colBlk + n;
            ((unsigned*)Wt[buf][n])[kp] = pack2(pw[0], pw[Cout]);
        }
    };

    v8f acc0 = {0.f,0.f,0.f,0.f,0.f,0.f,0.f,0.f};
    v8f acc1 = {0.f,0.f,0.f,0.f,0.f,0.f,0.f,0.f};
    const int m    = lane & 15;     // M row (A) / N col (B) within 16
    const int hf   = lane >> 4;     // lane half selects K sub-range
    const int nloc = wave * 16 + m; // column within the 64-wide block tile

    stage(0, 0);
    __syncthreads();
    for (int s = 0; s < KS; ++s) {
        const int buf = s & 1;
        if (s + 1 < KS) stage(s + 1, buf ^ 1);
        // CDNA5 16-bit WMMA layout: lane {idx, half}; K = hf*8+{0..7} (v0..3)
        // and 16+hf*8+{0..7} (v4..7) -> two contiguous 16B halves per fragment.
        v8bf a0l = *(const v8bf*)&As[buf][m][hf * 8];
        v8bf a0h = *(const v8bf*)&As[buf][m][hf * 8 + 16];
        v8bf a1l = *(const v8bf*)&As[buf][16 + m][hf * 8];
        v8bf a1h = *(const v8bf*)&As[buf][16 + m][hf * 8 + 16];
        v8bf bl  = *(const v8bf*)&Wt[buf][nloc][hf * 8];
        v8bf bh  = *(const v8bf*)&Wt[buf][nloc][hf * 8 + 16];
        v16bf a0 = combine(a0l, a0h);
        v16bf a1 = combine(a1l, a1h);
        v16bf bb = combine(bl, bh);
        acc0 = __builtin_amdgcn_wmma_f32_16x16x32_bf16(
                   false, a0, false, bb, (short)0, acc0, false, false);
        acc1 = __builtin_amdgcn_wmma_f32_16x16x32_bf16(
                   false, a1, false, bb, (short)0, acc1, false, false);
        __syncthreads();
    }

    // D layout: VGPR r, lanes 0-15 -> (M=r, N=lane); lanes 16-31 -> (M=r+8).
    // Store + fused BN statistics (sum/sumsq per output channel).
    float ssum = 0.f, ssq = 0.f;
#pragma unroll
    for (int r = 0; r < 8; ++r) {
        int mr = r + hf * 8;
        float v0 = acc0[r], v1 = acc1[r];
        Y[(long)(row0 + mr)      * Cout + colBlk + nloc] = v0;
        Y[(long)(row0 + 16 + mr) * Cout + colBlk + nloc] = v1;
        ssum += v0 + v1;
        ssq  += v0 * v0 + v1 * v1;
    }
    ssum += __shfl_xor(ssum, 16, 32);   // fold the two lane-halves (same col)
    ssq  += __shfl_xor(ssq,  16, 32);
    if (hf == 0) {
        atomicAdd(&stats[colBlk + nloc],        ssum);
        atomicAdd(&stats[Cout + colBlk + nloc], ssq);
    }
}

// ------------------------------ host sequencing ----------------------------

static int log2i(int v) { int s = 0; while ((1 << s) < v) ++s; return s; }

template <int Cin, int Cout>
static void launch_conv(hipStream_t st, const float* X, int Nv, const int* no,
                        const float* W, float* Y, float* stats) {
    long M = 64L * Nv;
    dim3 grid((unsigned)(M / 32), Cout / 64);
    conv_wmma_kernel<Cin, Cout><<<grid, 128, 0, st>>>(X, no, W, Y, Nv, stats);
}

static void conv_stage(hipStream_t st, const float* X, int Nv, int Cin,
                       const int* no, const float* W, int Cout, float* Y,
                       float* stats, const float* g, const float* be,
                       const float* res, int resC) {
    long M = 64L * Nv;
    zero_kernel<<<dim3(2), 256, 0, st>>>(stats, 2 * Cout);
    if      (Cin ==  32 && Cout ==  64) launch_conv< 32,  64>(st, X, Nv, no, W, Y, stats);
    else if (Cin ==  64 && Cout ==  64) launch_conv< 64,  64>(st, X, Nv, no, W, Y, stats);
    else if (Cin ==  64 && Cout == 128) launch_conv< 64, 128>(st, X, Nv, no, W, Y, stats);
    else if (Cin == 128 && Cout == 128) launch_conv<128, 128>(st, X, Nv, no, W, Y, stats);
    else if (Cin == 128 && Cout == 256) launch_conv<128, 256>(st, X, Nv, no, W, Y, stats);
    else                                launch_conv<256, 256>(st, X, Nv, no, W, Y, stats);
    int  cqShift = log2i(Cout) - 2;
    long totq    = M << cqShift;
    apply_bn_kernel<<<dim3((unsigned)((totq + 255) / 256)), 256, 0, st>>>(
        Y, M, cqShift, g, be, stats, res, resC);
}

static void res_block(hipStream_t st, void* const* din, int pb,
                      const float* xin, int Nv, int Cin, int C, const int* no,
                      float* h, float* out, float* stats) {
    // h = lrelu(bn(conv(x)))
    conv_stage(st, xin, Nv, Cin, no, (const float*)din[pb + 0], C, h, stats,
               (const float*)din[pb + 2], (const float*)din[pb + 3], nullptr, C);
    // out = lrelu(bn(conv(h)) + res)   (res concat-doubled when Cin == C/2)
    conv_stage(st, h, Nv, C, no, (const float*)din[pb + 4], C, out, stats,
               (const float*)din[pb + 6], (const float*)din[pb + 7], xin, Cin);
}

static void encode(hipStream_t st, const float* x,
                   float* out0, float* out1, float* out2, float* out3,
                   float* bufA, float* bufB, float* bufC, float* stats,
                   void* const* d_in) {
    const int* n10242 = (const int*)d_in[78];
    const int* n2562  = (const int*)d_in[79];
    const int* n642   = (const int*)d_in[80];
    const int* n162   = (const int*)d_in[81];

    // conv1 + BN + lrelu -> out0 (in d_out, in-place BN)
    long M0 = 64L * 10242;
    zero_kernel<<<dim3(1), 256, 0, st>>>(stats, 64);
    conv1_kernel<<<dim3((unsigned)((M0 + 255) / 256)), 256, 0, st>>>(
        x, n10242, (const float*)d_in[2], out0, 10242);
    stats32_kernel<<<dim3(240), 256, 0, st>>>(out0, M0, stats);
    apply_bn_kernel<<<dim3((unsigned)(((M0 << 3) + 255) / 256)), 256, 0, st>>>(
        out0, M0, 3, (const float*)d_in[4], (const float*)d_in[5], stats,
        nullptr, 32);

    // pool -> [64,2562,32]
    pool_kernel<<<dim3((unsigned)(((64L * 2562) << 3) + 255) / 256, 1), 256, 0, st>>>(
        out0, n10242, bufA, 10242, 2562, 3);
    // level 1 @ N=2562, 32 -> 64
    res_block(st, d_in, 6,  bufA, 2562, 32, 64, n2562, bufB, bufC, stats);
    res_block(st, d_in, 14, bufC, 2562, 64, 64, n2562, bufA, bufB, stats);
    res_block(st, d_in, 22, bufB, 2562, 64, 64, n2562, bufA, out1, stats);

    pool_kernel<<<dim3((unsigned)(((64L * 642) << 4) + 255) / 256, 1), 256, 0, st>>>(
        out1, n2562, bufA, 2562, 642, 4);
    // level 2 @ N=642, 64 -> 128
    res_block(st, d_in, 30, bufA, 642, 64, 128, n642, bufB, bufC, stats);
    res_block(st, d_in, 38, bufC, 642, 128, 128, n642, bufA, bufB, stats);
    res_block(st, d_in, 46, bufB, 642, 128, 128, n642, bufA, out2, stats);

    pool_kernel<<<dim3((unsigned)(((64L * 162) << 5) + 255) / 256, 1), 256, 0, st>>>(
        out2, n642, bufA, 642, 162, 5);
    // level 3 @ N=162, 128 -> 256
    res_block(st, d_in, 54, bufA, 162, 128, 256, n162, bufB, bufC, stats);
    res_block(st, d_in, 62, bufC, 162, 256, 256, n162, bufA, bufB, stats);
    res_block(st, d_in, 70, bufB, 162, 256, 256, n162, bufA, out3, stats);
}

extern "C" void kernel_launch(void* const* d_in, const int* in_sizes, int n_in,
                              void* d_out, int out_size, void* d_ws,
                              size_t ws_size, hipStream_t stream) {
    (void)in_sizes; (void)n_in; (void)out_size; (void)ws_size;

    float* out = (float*)d_out;
    const long S0 = 64L * 10242 * 32;
    const long S1 = 64L * 2562 * 64;
    const long S2 = 64L * 642 * 128;
    const long S3 = 64L * 162 * 256;
    float* lo0 = out;       float* lo1 = lo0 + S0;
    float* lo2 = lo1 + S1;  float* lo3 = lo2 + S2;
    float* ro0 = lo3 + S3;  float* ro1 = ro0 + S0;
    float* ro2 = ro1 + S1;  float* ro3 = ro2 + S2;

    // workspace: 3 ping-pong activation buffers (max level-1 size) + stats
    float* ws = (float*)d_ws;
    const long BUF = 64L * 2562 * 64;   // 10,493,952 floats
    float* bufA  = ws;
    float* bufB  = ws + BUF;
    float* bufC  = ws + 2 * BUF;
    float* stats = ws + 3 * BUF;        // 512 floats (sum | sumsq)

    encode(stream, (const float*)d_in[0], lo0, lo1, lo2, lo3,
           bufA, bufB, bufC, stats, d_in);
    encode(stream, (const float*)d_in[1], ro0, ro1, ro2, ro3,
           bufA, bufB, bufC, stats, d_in);
}
